// LSTMDP_3100966388315
// MI455X (gfx1250) — matrix operand order repaired and checked by reading, not compiled, not run on hardware
//
#include <hip/hip_runtime.h>
#include <stdint.h>

#define B_  64
#define T_  512
#define D_  128
#define H_  256

// packed weight fragment counts (bf16 elements)
#define NIH 131072   // W_ih: KT=4 tiles * NT=64 * (32 lanes * 16 elems)
#define NHH 262144   // W_hh: KT=8 tiles * NT=64 * 512

typedef __attribute__((ext_vector_type(16))) __bf16 v16bf;
typedef __attribute__((ext_vector_type(8)))  float  v8f;

// explicit global-address-space pointers -> global_load_b128 + SADDR form
typedef __attribute__((address_space(1))) const uint8_t* gbyte_p;
typedef __attribute__((address_space(1))) const v16bf*   gfrag_p;

__device__ __forceinline__ v8f wmma_bf16(v16bf a, v16bf b, v8f c) {
  // D = A(16x32 bf16) * B(32x16 bf16) + C(16x16 f32)
  return __builtin_amdgcn_wmma_f32_16x16x32_bf16(false, a, false, b, (short)0, c,
                                                 false, false);
}

__device__ __forceinline__ float sigm(float x) { return 1.0f / (1.0f + __expf(-x)); }
// fast tanh: 2*sigmoid(2x)-1; saturates correctly to +/-1 in fp32
__device__ __forceinline__ float tanh_fast(float x) {
  return 2.0f / (1.0f + __expf(-2.0f * x)) - 1.0f;
}

// weight B-fragment load: uniform base (SGPR pair) + 32-bit byte offset
__device__ __forceinline__ v16bf load_b_frag(gbyte_p base, uint32_t byteOff) {
  return *(gfrag_p)(base + byteOff);
}

// Build a 16x32 bf16 A-fragment from a row-major bf16 matrix in LDS.
// ISA 7.12.2 16-bit A layout: lane L -> M = L%16, half g = L/16;
//   VGPR v<4 : K = 2v + 8g (+1) ; VGPR v>=4: K = 16+2(v-4)+8g (+1)
// The 8 dword reads are two contiguous 16B runs -> 2x ds_load_b128.
__device__ __forceinline__ v16bf load_a_frag(const __bf16* base, int stride, int kt,
                                             int lane) {
  const int m  = lane & 15;
  const int g2 = lane >> 4;
  union { v16bf v; uint32_t u[8]; } r;
  const __bf16* row = base + m * stride + kt * 32 + 8 * g2;
#pragma unroll
  for (int v = 0; v < 8; ++v) {
    const int K0 = (v < 4) ? (2 * v) : (16 + 2 * (v - 4));
    r.u[v] = *reinterpret_cast<const uint32_t*>(row + K0);  // pair (K0, K0+1)
  }
  return r.v;
}

// GEMM for TWO gates (g0, g0+1), accumulating val/mu/r streams:
//   acc*(16x16) += h(16x256) x Whh-tiles + x(16x128) x Wih-tiles
// Weight fragments stream from L2 (1.5 MB resident) with GVS addressing:
// per-lane uint32 byte offset hoisted, kt contribution folds into imm offset.
__device__ __forceinline__ void ibp_gemm2(
    const __bf16* hS, int hRowStride, int hStreamStride,
    const __bf16* xS, int xRowStride, int xStreamStride,
    gbyte_p WhhW, gbyte_p WhhA, gbyte_p WihW, gbyte_p WihA,
    int g0, int wave, int lane,
    v8f accv[2], v8f accm[2], v8f accr[2]) {
  // fragment byte offsets: ((kt*64 + nt)*32 + lane) * 32 bytes
  const uint32_t off0 = (uint32_t)(((g0 + 0) * 16 + wave) * 1024 + lane * 32);
  const uint32_t off1 = (uint32_t)(((g0 + 1) * 16 + wave) * 1024 + lane * 32);
  // recurrent part: K = 256
#pragma unroll 2
  for (int kt = 0; kt < 8; ++kt) {
    const uint32_t ko = (uint32_t)kt * 65536u;   // kt*64*32lanes*32B
    const v16bf Av = load_a_frag(hS + 0 * hStreamStride, hRowStride, kt, lane);
    const v16bf Am = load_a_frag(hS + 1 * hStreamStride, hRowStride, kt, lane);
    const v16bf Ar = load_a_frag(hS + 2 * hStreamStride, hRowStride, kt, lane);
    const v16bf bw0 = load_b_frag(WhhW, off0 + ko);
    const v16bf ba0 = load_b_frag(WhhA, off0 + ko);
    const v16bf bw1 = load_b_frag(WhhW, off1 + ko);
    const v16bf ba1 = load_b_frag(WhhA, off1 + ko);
    accv[0] = wmma_bf16(Av, bw0, accv[0]);
    accm[0] = wmma_bf16(Am, bw0, accm[0]);
    accr[0] = wmma_bf16(Ar, ba0, accr[0]);
    accv[1] = wmma_bf16(Av, bw1, accv[1]);
    accm[1] = wmma_bf16(Am, bw1, accm[1]);
    accr[1] = wmma_bf16(Ar, ba1, accr[1]);
  }
  // input part (fused gx): K = 128
#pragma unroll 2
  for (int kt = 0; kt < 4; ++kt) {
    const uint32_t ko = (uint32_t)kt * 65536u;
    const v16bf Av = load_a_frag(xS + 0 * xStreamStride, xRowStride, kt, lane);
    const v16bf Am = load_a_frag(xS + 1 * xStreamStride, xRowStride, kt, lane);
    const v16bf Ar = load_a_frag(xS + 2 * xStreamStride, xRowStride, kt, lane);
    const v16bf bw0 = load_b_frag(WihW, off0 + ko);
    const v16bf ba0 = load_b_frag(WihA, off0 + ko);
    const v16bf bw1 = load_b_frag(WihW, off1 + ko);
    const v16bf ba1 = load_b_frag(WihA, off1 + ko);
    accv[0] = wmma_bf16(Av, bw0, accv[0]);
    accm[0] = wmma_bf16(Am, bw0, accm[0]);
    accr[0] = wmma_bf16(Ar, ba0, accr[0]);
    accv[1] = wmma_bf16(Av, bw1, accv[1]);
    accm[1] = wmma_bf16(Am, bw1, accm[1]);
    accr[1] = wmma_bf16(Ar, ba1, accr[1]);
  }
}

// ---------------------------------------------------------------------------
// Pack W_ih (1024x128) and W_hh (1024x256) + |W| variants into WMMA B-fragment
// order:  packed[((kt*64 + nt)*32 + lane)*16 + e] = bf16(W[n][k]) with
//   n = nt*16 + (lane&15),  k = kt*32 + 8*(lane>>4) + (e&7) + 16*(e>>3)   (ISA B layout)
// ---------------------------------------------------------------------------
__global__ void ibp_pack_weights(const float* __restrict__ Wih,
                                 const float* __restrict__ Whh,
                                 __bf16* __restrict__ ws) {
  const int total = 2 * NIH + 2 * NHH;
  for (int idx = blockIdx.x * blockDim.x + threadIdx.x; idx < total;
       idx += gridDim.x * blockDim.x) {
    const float* W; int K; bool ab; __bf16* dst; int p;
    if (idx < NIH)            { p = idx;               W = Wih; K = D_; ab = false; dst = ws; }
    else if (idx < 2*NIH)     { p = idx - NIH;         W = Wih; K = D_; ab = true;  dst = ws + NIH; }
    else if (idx < 2*NIH+NHH) { p = idx - 2*NIH;       W = Whh; K = H_; ab = false; dst = ws + 2*NIH; }
    else                      { p = idx - 2*NIH - NHH; W = Whh; K = H_; ab = true;  dst = ws + 2*NIH + NHH; }
    const int e    = p & 15;
    const int lane = (p >> 4) & 31;
    const int tile = p >> 9;
    const int nt   = tile & 63;
    const int kt   = tile >> 6;
    const int n    = nt * 16 + (lane & 15);
    const int k    = kt * 32 + ((lane >> 4) << 3) + (e & 7) + ((e >> 3) << 4);
    float v = W[(size_t)n * K + k];
    if (ab) v = fabsf(v);
    dst[p] = (__bf16)v;
  }
}

// ---------------------------------------------------------------------------
// Fused IBP-LSTM scan. One workgroup per 16-row batch chunk (grid = 4).
// 16 waves; wave w owns h-columns [w*16, w*16+16). h (val,mu,r) double-buffered
// in LDS as bf16; c (val,lb,ub) in per-wave v8f registers.
// Two gate passes per step (i,f then g,o). A zero-valued offset is laundered
// through an empty asm each step: keeps the weight loads inside the time loop
// (no LICM -> no spills) while preserving global address-space provenance.
// ---------------------------------------------------------------------------
__global__ __launch_bounds__(512, 1) void ibp_lstm_scan(
    const float* __restrict__ xv, const float* __restrict__ xl,
    const float* __restrict__ xu, const float* __restrict__ bias,
    const __bf16* __restrict__ pWihW, const __bf16* __restrict__ pWihA,
    const __bf16* __restrict__ pWhhW, const __bf16* __restrict__ pWhhA,
    float* __restrict__ out) {
  const int tid  = threadIdx.x;
  const int lane = tid & 31;
  const int wave = tid >> 5;            // 0..15 -> h-block
  const int b0   = blockIdx.x * 16;     // batch chunk base

  __shared__ __bf16 hb[2][3][16][H_ + 8];   // [buf][val/mu/r][row][col]
  __shared__ __bf16 xb[3][16][D_ + 8];      // x_t staged as val/mu/r

  const int HSS = 16 * (H_ + 8);            // h stream stride (elems)
  const int XSS = 16 * (D_ + 8);            // x stream stride (elems)

  for (int i = tid; i < 3 * HSS; i += 512)
    (&hb[0][0][0][0])[i] = (__bf16)0.0f;    // h0 = 0

  float bg[4];
#pragma unroll
  for (int g = 0; g < 4; ++g) bg[g] = bias[g * H_ + wave * 16 + (lane & 15)];

  v8f cv = {}, cl = {}, cu = {};            // c0 = 0

  size_t wofs = 0;                          // laundered each step (stays 0)

  __syncthreads();

  for (int t = 0; t < T_; ++t) {
    // Opaque zero offset: blocks LICM of weight loads across the time loop
    // without destroying the pointers' global-addrspace provenance.
    asm volatile("" : "+s"(wofs));
    const gbyte_p WIW = (gbyte_p)((const uint8_t*)pWihW + wofs);
    const gbyte_p WIA = (gbyte_p)((const uint8_t*)pWihA + wofs);
    const gbyte_p WHW = (gbyte_p)((const uint8_t*)pWhhW + wofs);
    const gbyte_p WHA = (gbyte_p)((const uint8_t*)pWhhA + wofs);

    // ---- stage x_t tile (16 x 128, fp32 -> bf16 val/mu/r) into LDS --------
    for (int i = tid; i < 16 * D_; i += 512) {
      const int row = i >> 7;
      const int col = i & (D_ - 1);
      const size_t gi = ((size_t)(b0 + row) * T_ + t) * D_ + col;
      const float v  = xv[gi];
      const float lo = xl[gi];
      const float hi = xu[gi];
      xb[0][row][col] = (__bf16)v;
      xb[1][row][col] = (__bf16)(0.5f * (lo + hi));
      xb[2][row][col] = (__bf16)(0.5f * (hi - lo));
      if (t + 1 < T_) {
        __builtin_prefetch(xv + gi + D_, 0, 0);
        __builtin_prefetch(xl + gi + D_, 0, 0);
        __builtin_prefetch(xu + gi + D_, 0, 0);
      }
    }
    __syncthreads();

    const __bf16* hS = &hb[t & 1][0][0][0];
    const __bf16* xS = &xb[0][0][0];

    // ================= PASS 1: gates i (0) and f (1) =======================
    v8f accv[2] = {}, accm[2] = {}, accr[2] = {};
    ibp_gemm2(hS, H_ + 8, HSS, xS, D_ + 8, XSS,
              WHW, WHA, WIW, WIA, /*g0=*/0, wave, lane, accv, accm, accr);

    v8f igv8, igl8, igu8;                  // keep input-gate activations live
#pragma unroll
    for (int e = 0; e < 8; ++e) {
      const float piv = accv[0][e] + bg[0], pil = accm[0][e] - accr[0][e] + bg[0], piu = accm[0][e] + accr[0][e] + bg[0];
      const float pfv = accv[1][e] + bg[1], pfl = accm[1][e] - accr[1][e] + bg[1], pfu = accm[1][e] + accr[1][e] + bg[1];
      igv8[e] = sigm(piv); igl8[e] = sigm(pil); igu8[e] = sigm(piu);
      const float fgv = sigm(pfv), fgl = sigm(pfl), fgu = sigm(pfu);
      // c <- f (x) c   (interval product; old c dies here)
      const float a1 = fgl * cl[e], a2 = fgl * cu[e], a3 = fgu * cl[e], a4 = fgu * cu[e];
      cv[e] = fgv * cv[e];
      cl[e] = fminf(fminf(a1, a2), fminf(a3, a4));
      cu[e] = fmaxf(fmaxf(a1, a2), fmaxf(a3, a4));
    }

    // ================= PASS 2: gates g (2) and o (3) =======================
#pragma unroll
    for (int q = 0; q < 2; ++q) { accv[q] = v8f{}; accm[q] = v8f{}; accr[q] = v8f{}; }
    ibp_gemm2(hS, H_ + 8, HSS, xS, D_ + 8, XSS,
              WHW, WHA, WIW, WIA, /*g0=*/2, wave, lane, accv, accm, accr);

    const int wb = (t + 1) & 1;
    const int n  = lane & 15;
    const int h  = wave * 16 + n;
#pragma unroll
    for (int e = 0; e < 8; ++e) {
      const float pgv = accv[0][e] + bg[2], pgl = accm[0][e] - accr[0][e] + bg[2], pgu = accm[0][e] + accr[0][e] + bg[2];
      const float pov = accv[1][e] + bg[3], pol = accm[1][e] - accr[1][e] + bg[3], pou = accm[1][e] + accr[1][e] + bg[3];
      const float ggv = tanh_fast(pgv), ggl = tanh_fast(pgl), ggu = tanh_fast(pgu);
      const float ogv = sigm(pov),  ogl = sigm(pol),  ogu = sigm(pou);
      // c <- c + i (x) g
      const float d1 = igl8[e] * ggl, d2 = igl8[e] * ggu, d3 = igu8[e] * ggl, d4 = igu8[e] * ggu;
      const float cnv = cv[e] + igv8[e] * ggv;
      const float cnl = cl[e] + fminf(fminf(d1, d2), fminf(d3, d4));
      const float cnu = cu[e] + fmaxf(fmaxf(d1, d2), fmaxf(d3, d4));
      cv[e] = cnv; cl[e] = cnl; cu[e] = cnu;
      // h = o (x) tanh(c)
      const float tv = tanh_fast(cnv), tl = tanh_fast(cnl), tu = tanh_fast(cnu);
      const float e1 = ogl * tl, e2 = ogl * tu, e3 = ogu * tl, e4 = ogu * tu;
      const float hvv = ogv * tv;
      const float hll = fminf(fminf(e1, e2), fminf(e3, e4));
      const float huu = fmaxf(fmaxf(e1, e2), fmaxf(e3, e4));

      const int m = e + 8 * (lane >> 4);   // C/D layout: M = r + 8*(lane/16)
      const int b = b0 + m;
      // output ys: (3, B, T, H)
      out[(((size_t)0 * B_ + b) * T_ + t) * H_ + h] = hvv;
      out[(((size_t)1 * B_ + b) * T_ + t) * H_ + h] = hll;
      out[(((size_t)2 * B_ + b) * T_ + t) * H_ + h] = huu;
      // next-step h state as bf16 val / mu / r
      hb[wb][0][m][h] = (__bf16)hvv;
      hb[wb][1][m][h] = (__bf16)(0.5f * (hll + huu));
      hb[wb][2][m][h] = (__bf16)(0.5f * (huu - hll));
    }
    __syncthreads();
  }
}

extern "C" void kernel_launch(void* const* d_in, const int* in_sizes, int n_in,
                              void* d_out, int out_size, void* d_ws, size_t ws_size,
                              hipStream_t stream) {
  (void)in_sizes; (void)n_in; (void)out_size; (void)ws_size;
  const float* xv  = (const float*)d_in[0];
  const float* xl  = (const float*)d_in[1];
  const float* xu  = (const float*)d_in[2];
  const float* Wih = (const float*)d_in[3];
  const float* Whh = (const float*)d_in[4];
  const float* b   = (const float*)d_in[5];
  float* out = (float*)d_out;
  __bf16* ws = (__bf16*)d_ws;   // 786432 bf16 = 1.5 MB of packed weights

  hipLaunchKernelGGL(ibp_pack_weights, dim3(1536), dim3(512), 0, stream, Wih, Whh, ws);
  hipLaunchKernelGGL(ibp_lstm_scan, dim3(4), dim3(512), 0, stream,
                     xv, xl, xu, b,
                     ws, ws + NIH, ws + 2 * NIH, ws + 2 * NIH + NHH, out);
}
